// Kmeans_81956565942450
// MI455X (gfx1250) — matrix operand order repaired and checked by reading, not compiled, not run on hardware
//
#include <hip/hip_runtime.h>

typedef __attribute__((ext_vector_type(16))) _Float16 v16h;
typedef __attribute__((ext_vector_type(8)))  _Float16 v8h;
typedef __attribute__((ext_vector_type(8)))  float    v8f;

#define Bn 128
#define Hn 14
#define Wn 14
#define Cn 512
#define HWn (Hn*Wn)                 // 196
#define NPTS 512                    // points (=channels) per batch
#define NELEM (Bn*HWn*Cn)           // 12,845,056 floats

// ---------------------------------------------------------------------------
// Kernel 1: per-(b,c) max-pixel coordinates -> points (float2: (arg1, arg0))
// Threads over (b,c); consecutive lanes -> consecutive c -> fully coalesced.
// ---------------------------------------------------------------------------
__global__ void k_points(const float* __restrict__ in, float2* __restrict__ pts) {
  int idx = blockIdx.x * blockDim.x + threadIdx.x;   // 0 .. B*C-1
  if (idx >= Bn * Cn) return;
  int b = idx >> 9;            // / 512
  int c = idx & (Cn - 1);
  const float* p = in + (size_t)b * HWn * Cn + c;

  float colmax[Wn];
  #pragma unroll
  for (int w = 0; w < Wn; ++w) colmax[w] = -__builtin_inff();

  float bestRow = -__builtin_inff(); int arg0 = 0;
  #pragma unroll
  for (int h = 0; h < Hn; ++h) {
    float rowm = -__builtin_inff();
    #pragma unroll
    for (int w = 0; w < Wn; ++w) {
      float v = p[(h * Wn + w) * Cn];          // immediate offsets after unroll
      rowm = fmaxf(rowm, v);
      colmax[w] = fmaxf(colmax[w], v);
    }
    if (rowm > bestRow) { bestRow = rowm; arg0 = h; }   // strict > = first argmax
  }
  float bestCol = -__builtin_inff(); int arg1 = 0;
  #pragma unroll
  for (int w = 0; w < Wn; ++w)
    if (colmax[w] > bestCol) { bestCol = colmax[w]; arg1 = w; }

  pts[idx] = make_float2((float)arg1, (float)arg0);
}

// ---------------------------------------------------------------------------
// Kernel 2: 2-means per batch, one wave32 per batch.
// Per-iteration masked reduction with v_wmma_f32_16x16x32_f16:
//   A(16x32 f16): row0 = x, row1 = y, row2 = 1, rows 3..15 = 0   (STATIC)
//   B(32x16 f16): mask of the same 32 points replicated per column (DYNAMIC)
//   D = A*B + C  ->  D[0,*]=Sum m*x, D[1,*]=Sum m*y, D[2,*]=Sum m.
// 16 chained WMMAs reduce all 512 points. Coords<=13 and masks 0/1 are exact
// in f16; f32 accumulation is exact. Fragment builds are contiguous
// ds_load_b128's from a per-lane row pointer -- zero divergence in the loop.
// ---------------------------------------------------------------------------
__global__ void k_kmeans(const float2* __restrict__ pts, float* __restrict__ mask) {
  const int b    = blockIdx.x;
  const int lane = threadIdx.x;           // 0..31, full wave (EXEC all ones)
  const float2* P = pts + b * NPTS;

  // [0,512)=x  [512,1024)=y  [1024,1536)=ones  [1536,2048)=zeros
  __shared__ _Float16 s_coord[4 * NPTS] __attribute__((aligned(32)));
  __shared__ _Float16 s_m[NPTS]         __attribute__((aligned(32)));

  float px[16], py[16];
  #pragma unroll
  for (int j = 0; j < 16; ++j) {
    int pid = lane + 32 * j;
    float2 q = P[pid];
    px[j] = q.x; py[j] = q.y;
    s_coord[          pid] = (_Float16)q.x;
    s_coord[NPTS    + pid] = (_Float16)q.y;
    s_coord[2*NPTS  + pid] = (_Float16)1.f;
    s_coord[3*NPTS  + pid] = (_Float16)0.f;
  }

  // Totals Sx, Sy over all 512 points (wave32 xor-shuffle reduce)
  float Sx = 0.f, Sy = 0.f;
  #pragma unroll
  for (int j = 0; j < 16; ++j) { Sx += px[j]; Sy += py[j]; }
  #pragma unroll
  for (int off = 16; off; off >>= 1) {
    Sx += __shfl_xor(Sx, off);
    Sy += __shfl_xor(Sy, off);
  }

  // cent0 = points[:2] of this batch
  float2 p0 = P[0], p1 = P[1];
  float c0x = p0.x, c0y = p0.y, c1x = p1.x, c1y = p1.y;
  unsigned mbits = 0;

  // Per-lane fragment geometry (computed once, branch-free afterwards):
  //  A: lane holds row M=lane&15; halves 0..7 -> K=rB0..rB0+7, 8..15 -> K=rB0+8..rB0+15
  //  B: lane holds column lane&15; halves i -> row rB0+i
  const int row = lane & 15;
  const _Float16* aSrc =
      s_coord + ((row == 0) ? 0 : (row == 1) ? NPTS : (row == 2) ? 2*NPTS : 3*NPTS);
  const int k0  = (lane < 16) ? 0 : 8;
  const int rB0 = (lane < 16) ? 0 : 16;

  __syncthreads();

  for (int it = 0; it <= 10; ++it) {         // ITERS+1 = 11 assignment steps
    mbits = 0;
    #pragma unroll
    for (int j = 0; j < 16; ++j) {
      int pid = lane + 32 * j;
      float dx0 = px[j] - c0x, dy0 = py[j] - c0y;
      float dx1 = px[j] - c1x, dy1 = py[j] - c1y;
      float d0 = dx0 * dx0 + dy0 * dy0;
      float d1 = dx1 * dx1 + dy1 * dy1;
      int m = (d1 < d0) ? 1 : 0;             // argmin ties -> cluster 0
      mbits |= (unsigned)m << j;
      s_m[pid] = (_Float16)(float)m;
    }
    __syncthreads();

    v8f acc = {};
    #pragma unroll
    for (int ch = 0; ch < 16; ++ch) {        // 16 chunks x 32 points
      const int base = ch * 32;
      // A-fragment: two contiguous 8-half runs (16B each) -> ds_load_b128 x2
      v8h alo = *(const v8h*)(aSrc + base + rB0);
      v8h ahi = *(const v8h*)(aSrc + base + rB0 + 8);
      v16h A = __builtin_shufflevector(alo, ahi,
               0,1,2,3,4,5,6,7,8,9,10,11,12,13,14,15);
      // B-fragment: 16 contiguous halves (32B, 32B-aligned)
      v16h Bf = *(const v16h*)(s_m + base + rB0);
      acc = __builtin_amdgcn_wmma_f32_16x16x32_f16(
          /*neg_a=*/false, A, /*neg_b=*/false, Bf,
          /*c_mod=*/(short)0, acc, /*reuse_a=*/false, /*reuse_b=*/false);
    }
    // Any lane<16 holds D rows 0..7 in acc[0..7]; broadcast from lane 0.
    float S1x = __shfl(acc[0], 0);
    float S1y = __shfl(acc[1], 0);
    float N1  = __shfl(acc[2], 0);

    float n1 = fmaxf(N1, 1.f);
    float n0 = fmaxf((float)NPTS - N1, 1.f);
    // NOTE: reference swaps clusters each update: c0 <- mean(cluster 1), c1 <- mean(cluster 0)
    float nc0x = S1x / n1,        nc0y = S1y / n1;
    float nc1x = (Sx - S1x) / n0, nc1y = (Sy - S1y) / n0;
    c0x = nc0x; c0y = nc0y; c1x = nc1x; c1y = nc1y;
    __syncthreads();
  }

  #pragma unroll
  for (int j = 0; j < 16; ++j) {
    int pid = lane + 32 * j;
    mask[b * NPTS + pid] = (float)((mbits >> j) & 1u);
  }
}

// ---------------------------------------------------------------------------
// Kernel 3: apply mask, producing C0 and C1 in one streamed pass (float4).
// Input re-read hits L2 (51 MB < 192 MB); outputs stream 102.8 MB to HBM.
// ---------------------------------------------------------------------------
__global__ void k_apply(const float* __restrict__ in, const float* __restrict__ mask,
                        float* __restrict__ out0, float* __restrict__ out1) {
  size_t t = (size_t)blockIdx.x * blockDim.x + threadIdx.x;
  size_t e = t * 4;
  if (e >= (size_t)NELEM) return;
  int b = (int)(e / ((size_t)HWn * Cn));
  int c = (int)(e & (size_t)(Cn - 1));      // Cn divides the row stride

  const float4 v = *(const float4*)(in + e);
  const float4 m = *(const float4*)(mask + (size_t)b * Cn + c);
  float4 a0, a1;
  a1.x = (m.x != 0.f) ? v.x : 0.f;  a0.x = (m.x != 0.f) ? 0.f : v.x;
  a1.y = (m.y != 0.f) ? v.y : 0.f;  a0.y = (m.y != 0.f) ? 0.f : v.y;
  a1.z = (m.z != 0.f) ? v.z : 0.f;  a0.z = (m.z != 0.f) ? 0.f : v.z;
  a1.w = (m.w != 0.f) ? v.w : 0.f;  a0.w = (m.w != 0.f) ? 0.f : v.w;
  *(float4*)(out0 + e) = a0;
  *(float4*)(out1 + e) = a1;
}

// ---------------------------------------------------------------------------
extern "C" void kernel_launch(void* const* d_in, const int* in_sizes, int n_in,
                              void* d_out, int out_size, void* d_ws, size_t ws_size,
                              hipStream_t stream) {
  const float* in  = (const float*)d_in[0];
  float*       out = (float*)d_out;

  float2* pts  = (float2*)d_ws;                                   // 128*512*8B = 512 KB
  float*  mask = (float*)((char*)d_ws + (size_t)Bn * Cn * sizeof(float2)); // 256 KB

  k_points<<<(Bn * Cn) / 256, 256, 0, stream>>>(in, pts);
  k_kmeans<<<Bn, 32, 0, stream>>>(pts, mask);
  const int nvec = NELEM / 4;                                     // 3,211,264
  k_apply<<<(nvec + 255) / 256, 256, 0, stream>>>(in, mask, out, out + (size_t)NELEM);
}
